// AQExpertWrapper_46832323395779
// MI455X (gfx1250) — matrix operand order repaired and checked by reading, not compiled, not run on hardware
//
#include <hip/hip_runtime.h>
#include <stdint.h>

// ---------------- problem constants ----------------
#define NEXP   8
#define DM     1024
#define DFF    4096
#define NTOK   8192
#define TOPK   2
#define NPAIR  (NTOK * TOPK)     // 16384 token-expert pairs
#define TILE   128               // M and N tile per workgroup
#define KSTEP  32                // bf16 WMMA K
#define LDSK   40                // padded LDS row stride (elements) -> conflict-free b128 frag loads
#define MAXTM  (NPAIR / TILE)    // 128 worst-case M tiles per expert

typedef __attribute__((ext_vector_type(16))) __bf16 v16bf;
typedef __attribute__((ext_vector_type(8)))  __bf16 v8bf;
typedef __attribute__((ext_vector_type(4)))  __bf16 v4bf;
typedef __attribute__((ext_vector_type(8)))  float  v8f;
typedef __attribute__((ext_vector_type(4)))  int    v4i;

#define GAS __attribute__((address_space(1)))
#define LAS __attribute__((address_space(3)))

#if defined(__AMDGCN__) && __has_builtin(__builtin_amdgcn_global_load_async_to_lds_b128) && \
    __has_builtin(__builtin_amdgcn_s_wait_asynccnt)
#define HAVE_ASYNC 1
#else
#define HAVE_ASYNC 0
#endif

// Build a 16-element bf16 fragment from two contiguous 16-byte spans (2x b128 loads).
static __device__ inline v16bf make_frag(const __bf16* lo_p, const __bf16* hi_p) {
    v8bf lo = *(const v8bf*)lo_p;
    v8bf hi = *(const v8bf*)hi_p;
    v16bf r;
#pragma unroll
    for (int i = 0; i < 8; ++i) { r[i] = lo[i]; r[i + 8] = hi[i]; }
    return r;
}

static __device__ inline float gelu_tanh(float x) {
    // jax.nn.gelu default (approximate=True)
    float x3 = x * x * x;
    return 0.5f * x * (1.0f + tanhf(0.7978845608028654f * (x + 0.044715f * x3)));
}

// ---------------- setup kernels ----------------
__global__ __launch_bounds__(256) void init_kernel(float* __restrict__ out, int* __restrict__ ctrl) {
    size_t n4 = (size_t)NTOK * DM / 4;
    float4 z; z.x = z.y = z.z = z.w = 0.0f;
    float4* o4 = (float4*)out;
    for (size_t i = (size_t)blockIdx.x * blockDim.x + threadIdx.x; i < n4;
         i += (size_t)gridDim.x * blockDim.x)
        o4[i] = z;
    if (blockIdx.x == 0 && threadIdx.x < 24) ctrl[threadIdx.x] = 0;  // counts/offsets/cursor
}

__global__ __launch_bounds__(256) void route_count_kernel(const int* __restrict__ sel,
                                                          int* __restrict__ counts) {
    int i = blockIdx.x * 256 + threadIdx.x;
    if (i < NPAIR) atomicAdd(&counts[sel[i]], 1);
}

__global__ void scan_kernel(int* __restrict__ ctrl) {
    // ctrl: [0..7]=counts, [8..15]=offsets, [16..23]=cursor
    if (threadIdx.x == 0 && blockIdx.x == 0) {
        int acc = 0;
#pragma unroll
        for (int e = 0; e < NEXP; ++e) {
            ctrl[8 + e]  = acc;
            ctrl[16 + e] = acc;
            acc += ctrl[e];
        }
    }
}

__global__ __launch_bounds__(256) void route_fill_kernel(const int* __restrict__ sel,
                                                         const float* __restrict__ rw,
                                                         int* __restrict__ cursor,
                                                         int* __restrict__ rowtok,
                                                         float* __restrict__ rowweight) {
    int i = blockIdx.x * 256 + threadIdx.x;
    if (i < NPAIR) {
        int e = sel[i];
        int pos = atomicAdd(&cursor[e], 1);
        rowtok[pos]    = i >> 1;   // token id (TOPK = 2)
        rowweight[pos] = rw[i];
    }
}

// ---------------- GEMM1: h = gelu(X_gathered @ W1[e][:, chunk]) (bf16 out) ----------------
__global__ __launch_bounds__(256) void ffn1_kernel(
    const float* __restrict__ hs, const float* __restrict__ w1,
    const int* __restrict__ rowtok, const int* __restrict__ counts,
    const int* __restrict__ offsets, unsigned short* __restrict__ hbuf_us,
    int ffbase, int ffchunk) {
    __shared__ __bf16 As[2][TILE][LDSK];   // [M][K], double buffered
    __shared__ __bf16 Bs[2][TILE][LDSK];   // [N][K] (W1 slice transposed), double buffered

    const int e   = blockIdx.z;
    const int cnt = counts[e];
    const int tm  = blockIdx.x;
    if (tm * TILE >= cnt) return;
    const int off    = offsets[e];
    const int nbaseC = blockIdx.y * TILE;        // col within this ff chunk
    const int nbaseW = ffbase + nbaseC;          // col within W1

    const int tid  = threadIdx.x;
    const int lane = tid & 31, wid = tid >> 5;
    const int wm = wid & 3, wn = wid >> 2;       // 4x2 wave grid; wave tile 32(M) x 64(N)
    const int half = lane >> 4, l15 = lane & 15;

    const float* w1e = w1 + (size_t)e * DM * DFF;
    __bf16* hbuf = (__bf16*)hbuf_us;

    // Hoisted per-thread A row pointers (loop-invariant token gather).
    // Out-of-range rows are clamped: garbage rows only affect discarded outputs.
    const float* aptr[4];
#pragma unroll
    for (int it = 0; it < 4; ++it) {
        int row = (tid >> 3) + it * 32;
        int gr  = tm * TILE + row;
        int cgr = gr < cnt ? gr : (cnt - 1);
        int tok = rowtok[off + cgr];
        aptr[it] = hs + (size_t)tok * DM + (tid & 7) * 4;
    }
    const float* bptr = w1e + (size_t)(tid >> 5) * DFF + nbaseW + (tid & 31) * 4;

    auto stage = [&](int k0, int b) {
#pragma unroll
        for (int it = 0; it < 4; ++it) {
            int row = (tid >> 3) + it * 32;
            int kq  = (tid & 7) * 4;
            float4 v = *(const float4*)(aptr[it] + k0);
            v4bf p; p[0] = (__bf16)v.x; p[1] = (__bf16)v.y; p[2] = (__bf16)v.z; p[3] = (__bf16)v.w;
            *(v4bf*)&As[b][row][kq] = p;
        }
#pragma unroll
        for (int it = 0; it < 4; ++it) {
            int k = (tid >> 5) + it * 8;
            int n = (tid & 31) * 4;
            const float* p = bptr + (size_t)(k0 + it * 8) * DFF;
            if (it == 0) __builtin_prefetch(p + 2 * (size_t)KSTEP * DFF, 0, 0);
            float4 v = *(const float4*)p;
            Bs[b][n + 0][k] = (__bf16)v.x;
            Bs[b][n + 1][k] = (__bf16)v.y;
            Bs[b][n + 2][k] = (__bf16)v.z;
            Bs[b][n + 3][k] = (__bf16)v.w;
        }
    };

    v8f acc[2][4] = {};

    stage(0, 0);
    __syncthreads();
#pragma unroll 2
    for (int k0 = 0; k0 < DM; k0 += KSTEP) {
        const int cur = (k0 >> 5) & 1, nxt = cur ^ 1;
        if (k0 + KSTEP < DM) stage(k0 + KSTEP, nxt);   // overlap next-tile loads with WMMA

        v16bf afr[2], bfr[4];
#pragma unroll
        for (int fm = 0; fm < 2; ++fm) {
            const __bf16* ap = &As[cur][wm * 32 + fm * 16 + l15][0];
            afr[fm] = make_frag(ap + half * 8, ap + 16 + half * 8);
        }
#pragma unroll
        for (int fn = 0; fn < 4; ++fn) {
            const __bf16* bp = &Bs[cur][wn * 64 + fn * 16 + l15][0];
            bfr[fn] = make_frag(bp + half * 8, bp + 16 + half * 8);
        }
#pragma unroll
        for (int fm = 0; fm < 2; ++fm)
#pragma unroll
            for (int fn = 0; fn < 4; ++fn)
                acc[fm][fn] = __builtin_amdgcn_wmma_f32_16x16x32_bf16(
                    false, afr[fm], false, bfr[fn], (short)0, acc[fm][fn], false, false);
        __syncthreads();
    }

    // ---- GELU + store bf16 to h buffer ----
#pragma unroll
    for (int fm = 0; fm < 2; ++fm) {
        int mbase = wm * 32 + fm * 16 + half * 8;
#pragma unroll
        for (int r = 0; r < 8; ++r) {
            int gr = tm * TILE + mbase + r;
            if (gr < cnt) {
                size_t rowoff = (size_t)(off + gr) * ffchunk;
#pragma unroll
                for (int fn = 0; fn < 4; ++fn) {
                    int nl = wn * 64 + fn * 16 + l15;
                    hbuf[rowoff + nbaseC + nl] = (__bf16)gelu_tanh(acc[fm][fn][r]);
                }
            }
        }
    }
}

// ---------------- GEMM2: out += (h @ W2[e][chunk, :]) * weight[row] ----------------
__global__ __launch_bounds__(256) void ffn2_kernel(
    const unsigned short* __restrict__ hbuf_us, const float* __restrict__ w2,
    const int* __restrict__ rowtok, const float* __restrict__ rowweight,
    const int* __restrict__ counts, const int* __restrict__ offsets,
    float* __restrict__ out, int ffbase, int ffchunk) {
    __shared__ __bf16 Asl[2][TILE][LDSK];  // [M][K] (bf16 h tile), double buffered
    __shared__ __bf16 Bs[2][TILE][LDSK];   // [N][K] (W2 slice transposed), double buffered

    const int e   = blockIdx.z;
    const int cnt = counts[e];
    const int tm  = blockIdx.x;
    if (tm * TILE >= cnt) return;
    const int off   = offsets[e];
    const int nbase = blockIdx.y * TILE;     // output column base

    const int tid  = threadIdx.x;
    const int lane = tid & 31, wid = tid >> 5;
    const int wm = wid & 3, wn = wid >> 2;
    const int half = lane >> 4, l15 = lane & 15;

    const __bf16* hbuf = (const __bf16*)hbuf_us;
    const float*  w2e  = w2 + (size_t)e * DFF * DM;

    // Hoisted A-tile copy source pointers: tile is 128 rows x 32 cols bf16 = 512 x 16B chunks,
    // 2 chunks per thread.  Out-of-range rows clamped (discarded at the epilogue).
    const __bf16* asrc[2];
#pragma unroll
    for (int it = 0; it < 2; ++it) {
        int c = tid + it * 256;
        int row = c >> 2, part = c & 3;
        int gr  = tm * TILE + row;
        int cgr = gr < cnt ? gr : (cnt - 1);
        asrc[it] = hbuf + (size_t)(off + cgr) * ffchunk + part * 8;
    }
    const float* bptr = w2e + (size_t)(ffbase + (tid >> 5)) * DM + nbase + (tid & 31) * 4;

    auto stage = [&](int k0, int b) {
#pragma unroll
        for (int it = 0; it < 2; ++it) {
            int c = tid + it * 256;
            int row = c >> 2, part = c & 3;
            __bf16* dst = &Asl[b][row][part * 8];
#if HAVE_ASYNC
            // pure bf16 byte-copy -> CDNA5 async global->LDS path (ASYNCcnt)
            __builtin_amdgcn_global_load_async_to_lds_b128(
                (GAS v4i*)(uintptr_t)(const void*)(asrc[it] + k0),
                (LAS v4i*)(uint32_t)(uintptr_t)(void*)dst, 0, 0);
#else
            *(v8bf*)dst = *(const v8bf*)(asrc[it] + k0);
#endif
        }
#pragma unroll
        for (int it = 0; it < 4; ++it) {
            int k = (tid >> 5) + it * 8;
            int n = (tid & 31) * 4;
            const float* p = bptr + (size_t)(k0 + it * 8) * DM;
            if (it == 0) __builtin_prefetch(p + 2 * (size_t)KSTEP * DM, 0, 0);
            float4 v = *(const float4*)p;
            Bs[b][n + 0][k] = (__bf16)v.x;
            Bs[b][n + 1][k] = (__bf16)v.y;
            Bs[b][n + 2][k] = (__bf16)v.z;
            Bs[b][n + 3][k] = (__bf16)v.w;
        }
    };

    v8f acc[2][4] = {};

    stage(0, 0);
#if HAVE_ASYNC
    __builtin_amdgcn_s_wait_asynccnt(0);
#endif
    __syncthreads();
#pragma unroll 2
    for (int k0 = 0; k0 < ffchunk; k0 += KSTEP) {
        const int cur = (k0 >> 5) & 1, nxt = cur ^ 1;
        if (k0 + KSTEP < ffchunk) stage(k0 + KSTEP, nxt);

        v16bf afr[2], bfr[4];
#pragma unroll
        for (int fm = 0; fm < 2; ++fm) {
            const __bf16* ap = &Asl[cur][wm * 32 + fm * 16 + l15][0];
            afr[fm] = make_frag(ap + half * 8, ap + 16 + half * 8);
        }
#pragma unroll
        for (int fn = 0; fn < 4; ++fn) {
            const __bf16* bp = &Bs[cur][wn * 64 + fn * 16 + l15][0];
            bfr[fn] = make_frag(bp + half * 8, bp + 16 + half * 8);
        }
#pragma unroll
        for (int fm = 0; fm < 2; ++fm)
#pragma unroll
            for (int fn = 0; fn < 4; ++fn)
                acc[fm][fn] = __builtin_amdgcn_wmma_f32_16x16x32_bf16(
                    false, afr[fm], false, bfr[fn], (short)0, acc[fm][fn], false, false);
#if HAVE_ASYNC
        __builtin_amdgcn_s_wait_asynccnt(0);
#endif
        __syncthreads();
    }

    // ---- weighted combine into output (two experts may hit a token -> f32 atomics) ----
#pragma unroll
    for (int fm = 0; fm < 2; ++fm) {
        int mbase = wm * 32 + fm * 16 + half * 8;
#pragma unroll
        for (int r = 0; r < 8; ++r) {
            int gr = tm * TILE + mbase + r;
            if (gr < cnt) {
                int hr = off + gr;
                int tok = rowtok[hr];
                float wgt = rowweight[hr];
#pragma unroll
                for (int fn = 0; fn < 4; ++fn) {
                    int nl = nbase + wn * 64 + fn * 16 + l15;
                    atomicAdd(out + (size_t)tok * DM + nl, acc[fm][fn][r] * wgt);
                }
            }
        }
    }
}

// ---------------- launch ----------------
extern "C" void kernel_launch(void* const* d_in, const int* in_sizes, int n_in,
                              void* d_out, int out_size, void* d_ws, size_t ws_size,
                              hipStream_t stream) {
    const float* hs  = (const float*)d_in[0];   // [8192,1024]
    const int*   sel = (const int*)d_in[1];     // [8192,2]
    const float* rw  = (const float*)d_in[2];   // [8192,2]
    const float* w1  = (const float*)d_in[3];   // [8,1024,4096]
    const float* w2  = (const float*)d_in[4];   // [8,4096,1024]
    float* out = (float*)d_out;

    // workspace layout
    uint8_t* w = (uint8_t*)d_ws;
    int* ctrl = (int*)w;                               // counts[8], offsets[8], cursor[8]
    int* rowtok = ctrl + 24;                           // [16384]
    float* rowweight = (float*)(rowtok + NPAIR);       // [16384]
    uintptr_t hb = ((uintptr_t)(rowweight + NPAIR) + 255) & ~(uintptr_t)255;
    unsigned short* hbuf = (unsigned short*)hb;        // [16384, ffchunk] bf16

    // pick largest ff chunk that fits the workspace (>=128)
    size_t fixed = (hb - (uintptr_t)w);
    int ffchunk = 512;
    while (ffchunk > 128 && fixed + (size_t)NPAIR * ffchunk * 2 > ws_size) ffchunk >>= 1;

    init_kernel<<<2048, 256, 0, stream>>>(out, ctrl);
    route_count_kernel<<<NPAIR / 256, 256, 0, stream>>>(sel, ctrl);
    scan_kernel<<<1, 32, 0, stream>>>(ctrl);
    route_fill_kernel<<<NPAIR / 256, 256, 0, stream>>>(sel, rw, ctrl + 16, rowtok, rowweight);

    dim3 g1(MAXTM, ffchunk / TILE, NEXP);
    dim3 g2(MAXTM, DM / TILE, NEXP);
    for (int ffbase = 0; ffbase < DFF; ffbase += ffchunk) {
        ffn1_kernel<<<g1, 256, 0, stream>>>(hs, w1, rowtok, ctrl, ctrl + 8, hbuf, ffbase, ffchunk);
        ffn2_kernel<<<g2, 256, 0, stream>>>(hbuf, w2, rowtok, rowweight, ctrl, ctrl + 8, out,
                                            ffbase, ffchunk);
    }
}
